// MappingNetwork_85169201479770
// MI455X (gfx1250) — compile-verified, matrix operand order checked by reading, not run on hardware
//
#include <hip/hip_runtime.h>

#define BATCH 16384
#define NZ 64
#define NDOM 8
#define SDIM 128
#define HID 512
#define PADB (BATCH + NDOM * 128)   // 17408 padded rows (groups padded to 128)
#define MTILES_PB (PADB / 128)      // 136 row tiles

typedef __attribute__((ext_vector_type(16))) _Float16 v16h;
typedef __attribute__((ext_vector_type(8)))  _Float16 v8h;
typedef __attribute__((ext_vector_type(8)))  float    v8f;
typedef __attribute__((ext_vector_type(4)))  int      v4i;

#if __has_builtin(__builtin_amdgcn_global_load_async_to_lds_b128)
#define HAS_ASYNC_LDS 1
#define GLB_V4I(p) ((__attribute__((address_space(1))) v4i*)(p))
#define LDS_V4I(p) ((__attribute__((address_space(3))) v4i*)(p))
#else
#define HAS_ASYNC_LDS 0
#endif

__device__ __forceinline__ void wait_async0() {
#if __has_builtin(__builtin_amdgcn_s_wait_asynccnt)
    __builtin_amdgcn_s_wait_asynccnt(0);
#else
    asm volatile("s_wait_asynccnt 0x0" ::: "memory");
#endif
}

// ---------------------------------------------------------------------------
// Weight convert + transpose:  f32 [K x N]  ->  f16 [N x K]   (per domain z)
// ---------------------------------------------------------------------------
__global__ __launch_bounds__(256) void wcvt_t(const float* __restrict__ in,
                                              _Float16* __restrict__ outp,
                                              int K, int N) {
    __shared__ float t[32][33];
    in   += (long)blockIdx.z * K * N;
    outp += (long)blockIdx.z * N * K;
    int k0 = blockIdx.y * 32, n0 = blockIdx.x * 32;
    int tx = threadIdx.x & 31, ty = threadIdx.x >> 5;
#pragma unroll
    for (int r = ty; r < 32; r += 8)
        t[r][tx] = in[(long)(k0 + r) * N + n0 + tx];
    __syncthreads();
#pragma unroll
    for (int r = ty; r < 32; r += 8)
        outp[(long)(n0 + r) * K + k0 + tx] = (_Float16)t[tx][r];
}

// ---------------------------------------------------------------------------
// Tiled WMMA GEMM:  Y = act(X @ W[dom] + b[dom])
//   X:  [M x K] f16 row-major;  Wt: [N x K] f16 (pre-transposed)
//   Block tile 128x128, BK=32, double-buffered LDS, 8 waves (2x4).
//   K, N compile-time for shift-only addressing.
// ---------------------------------------------------------------------------
template <bool RELU, bool OUTF32, int K, int N>
__global__ __launch_bounds__(256) void gemm_wmma(
    const _Float16* __restrict__ X, const _Float16* __restrict__ Wall,
    const float* __restrict__ ball, void* __restrict__ Yv,
    const int* __restrict__ tile2dom, long wstride, long bstride)
{
    __shared__ _Float16 As[2][128 * 40];   // row-major, padded stride 40
    __shared__ _Float16 Bs[2][128 * 40];   // rows of Wt (= cols of W)

    const int tid    = threadIdx.x;
    const int lane   = tid & 31;
    const int wid    = tid >> 5;
    const int wave_m = wid >> 2;   // 0..1  -> 64-row strip
    const int wave_n = wid & 3;    // 0..3  -> 32-col strip
    const int gm0    = blockIdx.y * 128;
    const int gn0    = blockIdx.x * 128;

    const int dom = tile2dom ? tile2dom[blockIdx.y] : 0;
    const _Float16* Wd = Wall + (long)dom * wstride;
    const float*    bd = ball + (long)dom * bstride;

    v8f acc[4][2];
#pragma unroll
    for (int i = 0; i < 4; i++)
#pragma unroll
        for (int j = 0; j < 2; j++) acc[i][j] = (v8f){};

    const int l15   = lane & 15;
    const int khalf = (lane >> 4) * 8;

    // staging assignment: 2 threads per row/col, 16 halves each
    const int srow = tid >> 1;
    const int shof = (tid & 1) * 16;
    const _Float16* aSrc = X  + (long)(gm0 + srow) * K + shof;
    const _Float16* bSrc = Wd + (long)(gn0 + srow) * K + shof;

    constexpr int nsteps = K >> 5;

#if HAS_ASYNC_LDS
    auto issue_stage = [&](int s, int b) {
        const _Float16* ga = aSrc + s * 32;
        const _Float16* gb = bSrc + s * 32;
        _Float16* la = &As[b][srow * 40 + shof];
        _Float16* lb = &Bs[b][srow * 40 + shof];
        __builtin_amdgcn_global_load_async_to_lds_b128(GLB_V4I(ga),     LDS_V4I(la),     0, 0);
        __builtin_amdgcn_global_load_async_to_lds_b128(GLB_V4I(ga + 8), LDS_V4I(la + 8), 0, 0);
        __builtin_amdgcn_global_load_async_to_lds_b128(GLB_V4I(gb),     LDS_V4I(lb),     0, 0);
        __builtin_amdgcn_global_load_async_to_lds_b128(GLB_V4I(gb + 8), LDS_V4I(lb + 8), 0, 0);
    };
    issue_stage(0, 0);
#else
    v8h ar0, ar1, br0, br1;
    auto load_stage = [&](int s) {
        const v8h* pa = (const v8h*)(aSrc + s * 32);
        ar0 = pa[0]; ar1 = pa[1];
        const v8h* pb = (const v8h*)(bSrc + s * 32);
        br0 = pb[0]; br1 = pb[1];
    };
    auto store_stage = [&](int b) {
        *(v8h*)(&As[b][srow * 40 + shof])     = ar0;
        *(v8h*)(&As[b][srow * 40 + shof + 8]) = ar1;
        *(v8h*)(&Bs[b][srow * 40 + shof])     = br0;
        *(v8h*)(&Bs[b][srow * 40 + shof + 8]) = br1;
    };
    load_stage(0);
    store_stage(0);
#endif

    for (int s = 0; s < nsteps; s++) {
        const int buf = s & 1;
#if HAS_ASYNC_LDS
        wait_async0();                         // stage s landed in LDS
        __syncthreads();                       // published to all waves
        if (s + 1 < nsteps) issue_stage(s + 1, buf ^ 1);  // DMA under compute
#else
        __syncthreads();                       // buf s&1 ready
        if (s + 1 < nsteps) load_stage(s + 1); // prefetch under compute
#endif

        v16h afr[4], bfr[2];
#pragma unroll
        for (int i = 0; i < 4; i++) {
            const _Float16* p = &As[buf][(wave_m * 64 + i * 16 + l15) * 40];
            v8h lo = *(const v8h*)(p + khalf);
            v8h hi = *(const v8h*)(p + 16 + khalf);
            afr[i] = __builtin_shufflevector(lo, hi, 0,1,2,3,4,5,6,7,8,9,10,11,12,13,14,15);
        }
#pragma unroll
        for (int j = 0; j < 2; j++) {
            const _Float16* p = &Bs[buf][(wave_n * 32 + j * 16 + l15) * 40];
            v8h lo = *(const v8h*)(p + khalf);
            v8h hi = *(const v8h*)(p + 16 + khalf);
            bfr[j] = __builtin_shufflevector(lo, hi, 0,1,2,3,4,5,6,7,8,9,10,11,12,13,14,15);
        }
#pragma unroll
        for (int i = 0; i < 4; i++)
#pragma unroll
            for (int j = 0; j < 2; j++)
                acc[i][j] = __builtin_amdgcn_wmma_f32_16x16x32_f16(
                    false, afr[i], false, bfr[j], (short)0, acc[i][j], false, false);

#if !HAS_ASYNC_LDS
        if (s + 1 < nsteps) {
            __syncthreads();                   // prior reads of buf^1 done
            store_stage(buf ^ 1);
        }
#endif
    }

    // ---- epilogue: bias (+ReLU), constant-offset stores ----
#pragma unroll
    for (int i = 0; i < 4; i++) {
#pragma unroll
        for (int j = 0; j < 2; j++) {
            const int n  = gn0 + wave_n * 32 + j * 16 + l15;
            const int m0 = gm0 + wave_m * 64 + i * 16 + (lane >> 4) * 8;
            const float bia = bd[n];
            if (OUTF32) {
                float* yp = (float*)Yv + (long)m0 * N + n;
#pragma unroll
                for (int v = 0; v < 8; v++) {
                    float r = acc[i][j][v] + bia;
                    if (RELU) r = fmaxf(r, 0.f);
                    yp[(long)v * N] = r;
                }
            } else {
                _Float16* yp = (_Float16*)Yv + (long)m0 * N + n;
#pragma unroll
                for (int v = 0; v < 8; v++) {
                    float r = acc[i][j][v] + bia;
                    if (RELU) r = fmaxf(r, 0.f);
                    yp[(long)v * N] = (_Float16)r;
                }
            }
        }
    }
}

// ---------------------------------------------------------------------------
// Bookkeeping kernels (counting sort by domain, padded to 128-row tiles)
// ---------------------------------------------------------------------------
__global__ void cvt_f32_f16(const float* __restrict__ x, _Float16* __restrict__ y, int n) {
    int i = blockIdx.x * 256 + threadIdx.x;
    if (i < n) y[i] = (_Float16)x[i];
}

__global__ void count_dom(const int* __restrict__ y, int* __restrict__ counts, int n) {
    int i = blockIdx.x * 256 + threadIdx.x;
    if (i < n) atomicAdd(&counts[y[i]], 1);
}

__global__ void scan_meta(const int* __restrict__ counts, int* __restrict__ pos,
                          int* __restrict__ tile2dom) {
    if (blockIdx.x == 0 && threadIdx.x == 0) {
        int start = 0;
        for (int d = 0; d < NDOM; d++) {
            pos[d] = start;
            int tiles = (counts[d] + 127) >> 7;
            int t0 = start >> 7;
            for (int t = 0; t < tiles; t++) tile2dom[t0 + t] = d;
            start += tiles << 7;
        }
        for (int t = start >> 7; t < MTILES_PB; t++) tile2dom[t] = 0;
    }
}

__global__ void scatter_perm(const int* __restrict__ y, int* __restrict__ pos,
                             int* __restrict__ perm, int n) {
    int i = blockIdx.x * 256 + threadIdx.x;
    if (i < n) {
        int p = atomicAdd(&pos[y[i]], 1);
        perm[p] = i;
    }
}

__global__ void gather_rows(const _Float16* __restrict__ h, const int* __restrict__ perm,
                            _Float16* __restrict__ hg) {
    int p = blockIdx.x;
    int src = perm[p];
    uint4* dst = (uint4*)(hg + (long)p * HID);
    if (src >= 0) {
        const uint4* s = (const uint4*)(h + (long)src * HID);
        dst[threadIdx.x] = s[threadIdx.x];
    } else {
        dst[threadIdx.x] = make_uint4(0u, 0u, 0u, 0u);
    }
}

__global__ void scatter_out(const float* __restrict__ sp, const int* __restrict__ perm,
                            float* __restrict__ out) {
    int p = blockIdx.x;
    int dstrow = perm[p];
    if (dstrow >= 0)
        out[(long)dstrow * SDIM + threadIdx.x] = sp[(long)p * SDIM + threadIdx.x];
}

// ---------------------------------------------------------------------------
extern "C" void kernel_launch(void* const* d_in, const int* in_sizes, int n_in,
                              void* d_out, int out_size, void* d_ws, size_t ws_size,
                              hipStream_t stream) {
    const float* z  = (const float*)d_in[0];
    const int*   y  = (const int*)d_in[1];
    const float* W0 = (const float*)d_in[2];  const float* b0 = (const float*)d_in[3];
    const float* W1 = (const float*)d_in[4];  const float* b1 = (const float*)d_in[5];
    const float* W2 = (const float*)d_in[6];  const float* b2 = (const float*)d_in[7];
    const float* W3 = (const float*)d_in[8];  const float* b3 = (const float*)d_in[9];
    const float* U1 = (const float*)d_in[10]; const float* c1 = (const float*)d_in[11];
    const float* U2 = (const float*)d_in[12]; const float* c2 = (const float*)d_in[13];
    const float* U3 = (const float*)d_in[14]; const float* c3 = (const float*)d_in[15];
    const float* Uo = (const float*)d_in[16]; const float* co = (const float*)d_in[17];
    float* out = (float*)d_out;

    char* ws = (char*)d_ws;
    size_t off = 0;
    auto carve = [&](size_t bytes) {
        void* p = ws + off;
        off += (bytes + 255) & ~(size_t)255;
        return p;
    };
    _Float16* zh = (_Float16*)carve((size_t)BATCH * NZ * 2);
    _Float16* P0 = (_Float16*)carve((size_t)PADB * HID * 2);
    _Float16* P1 = (_Float16*)carve((size_t)PADB * HID * 2);
    _Float16* P2 = (_Float16*)carve((size_t)PADB * HID * 2);
    float*  OutP = (float*)carve((size_t)PADB * SDIM * 4);
    _Float16* W0t = (_Float16*)carve((size_t)NZ * HID * 2);
    _Float16* W1t = (_Float16*)carve((size_t)HID * HID * 2);
    _Float16* W2t = (_Float16*)carve((size_t)HID * HID * 2);
    _Float16* W3t = (_Float16*)carve((size_t)HID * HID * 2);
    _Float16* U1t = (_Float16*)carve((size_t)NDOM * HID * HID * 2);
    _Float16* U2t = (_Float16*)carve((size_t)NDOM * HID * HID * 2);
    _Float16* U3t = (_Float16*)carve((size_t)NDOM * HID * HID * 2);
    _Float16* Uot = (_Float16*)carve((size_t)NDOM * HID * SDIM * 2);
    int* counts   = (int*)carve(NDOM * 4);
    int* pos      = (int*)carve(NDOM * 4);
    int* tile2dom = (int*)carve(MTILES_PB * 4);
    int* perm     = (int*)carve((size_t)PADB * 4);

    (void)hipMemsetAsync(counts, 0, NDOM * 4, stream);
    (void)hipMemsetAsync(perm, 0xFF, (size_t)PADB * 4, stream);   // -1 = padding row

    count_dom<<<BATCH / 256, 256, 0, stream>>>(y, counts, BATCH);
    scan_meta<<<1, 32, 0, stream>>>(counts, pos, tile2dom);
    scatter_perm<<<BATCH / 256, 256, 0, stream>>>(y, pos, perm, BATCH);

    cvt_f32_f16<<<(BATCH * NZ + 255) / 256, 256, 0, stream>>>(z, zh, BATCH * NZ);
    wcvt_t<<<dim3(HID / 32, NZ / 32, 1),     256, 0, stream>>>(W0, W0t, NZ,  HID);
    wcvt_t<<<dim3(HID / 32, HID / 32, 1),    256, 0, stream>>>(W1, W1t, HID, HID);
    wcvt_t<<<dim3(HID / 32, HID / 32, 1),    256, 0, stream>>>(W2, W2t, HID, HID);
    wcvt_t<<<dim3(HID / 32, HID / 32, 1),    256, 0, stream>>>(W3, W3t, HID, HID);
    wcvt_t<<<dim3(HID / 32, HID / 32, NDOM), 256, 0, stream>>>(U1, U1t, HID, HID);
    wcvt_t<<<dim3(HID / 32, HID / 32, NDOM), 256, 0, stream>>>(U2, U2t, HID, HID);
    wcvt_t<<<dim3(HID / 32, HID / 32, NDOM), 256, 0, stream>>>(U3, U3t, HID, HID);
    wcvt_t<<<dim3(SDIM / 32, HID / 32, NDOM),256, 0, stream>>>(Uo, Uot, HID, SDIM);

    dim3 blk(256);
    gemm_wmma<true, false, NZ, HID><<<dim3(HID / 128, BATCH / 128), blk, 0, stream>>>(
        zh, W0t, b0, P0, nullptr, 0, 0);
    gemm_wmma<true, false, HID, HID><<<dim3(HID / 128, BATCH / 128), blk, 0, stream>>>(
        P0, W1t, b1, P1, nullptr, 0, 0);
    gemm_wmma<true, false, HID, HID><<<dim3(HID / 128, BATCH / 128), blk, 0, stream>>>(
        P1, W2t, b2, P0, nullptr, 0, 0);
    gemm_wmma<true, false, HID, HID><<<dim3(HID / 128, BATCH / 128), blk, 0, stream>>>(
        P0, W3t, b3, P1, nullptr, 0, 0);

    gather_rows<<<PADB, 64, 0, stream>>>(P1, perm, P2);

    gemm_wmma<true, false, HID, HID><<<dim3(HID / 128, MTILES_PB), blk, 0, stream>>>(
        P2, U1t, c1, P0, tile2dom, (long)HID * HID, HID);
    gemm_wmma<true, false, HID, HID><<<dim3(HID / 128, MTILES_PB), blk, 0, stream>>>(
        P0, U2t, c2, P2, tile2dom, (long)HID * HID, HID);
    gemm_wmma<true, false, HID, HID><<<dim3(HID / 128, MTILES_PB), blk, 0, stream>>>(
        P2, U3t, c3, P0, tile2dom, (long)HID * HID, HID);
    gemm_wmma<false, true, HID, SDIM><<<dim3(SDIM / 128, MTILES_PB), blk, 0, stream>>>(
        P0, Uot, co, OutP, tile2dom, (long)HID * SDIM, SDIM);

    scatter_out<<<PADB, SDIM, 0, stream>>>(OutP, perm, out);
}
